// Q_MAB_78795470012895
// MI455X (gfx1250) — matrix operand order repaired and checked by reading, not compiled, not run on hardware
//
#include <hip/hip_runtime.h>
#include <hip/hip_bf16.h>

typedef __attribute__((ext_vector_type(16))) _Float16 v16h;
typedef __attribute__((ext_vector_type(8)))  _Float16 v8h;
typedef __attribute__((ext_vector_type(8)))  float    v8f;
typedef __attribute__((ext_vector_type(4)))  int      v4i;

#define B_    8
#define LQ_   1024
#define LK_   1024
#define QIN_  256
#define D_    512
#define H_    8
#define DK_   64
#define MROWS (B_ * LQ_)   // 8192

// ---------------------------------------------------------------------------
// Async global->LDS copy (CDNA5). 16 bytes per call per lane.
// Builtin signature (from clang diagnostics): arg0 = int4 addrspace(1)*,
// arg1 = LDS pointer, then imm offset and imm cpol.
// ---------------------------------------------------------------------------
#if defined(__HIP_DEVICE_COMPILE__) && defined(__gfx1250__) &&         \
    __has_builtin(__builtin_amdgcn_global_load_async_to_lds_b128) &&   \
    __has_builtin(__builtin_amdgcn_s_wait_asynccnt)
#define HAVE_ASYNC_LDS 1
#else
#define HAVE_ASYNC_LDS 0
#endif

typedef __attribute__((address_space(1))) v4i as1_v4i;
typedef __attribute__((address_space(3))) v4i as3_v4i;

__device__ __forceinline__ void async_cp16(const _Float16* g, _Float16* l) {
#if HAVE_ASYNC_LDS
  __builtin_amdgcn_global_load_async_to_lds_b128(
      (as1_v4i*)g, (as3_v4i*)l, 0, 0);
#else
  *(v8h*)l = *(const v8h*)g;
#endif
}

#if HAVE_ASYNC_LDS
#define WAIT_ASYNC_LE(n) __builtin_amdgcn_s_wait_asynccnt(n)
#else
#define WAIT_ASYNC_LE(n) ((void)0)
#endif

// ---------------------------------------------------------------------------
// Fragment loaders (layouts per CDNA5 ISA 7.12.2, wave32, f16 16x16x32)
// A (16x32, MxK): lane L -> M=L&15, g=L>>4; a[0..7]=A[M][c0+g*8+j],
//                 a[8..15]=A[M][c0+16+g*8+j]
// ---------------------------------------------------------------------------
__device__ __forceinline__ v16h load_fragA(const _Float16* __restrict__ base,
                                           int ld, int r0, int c0) {
  const int L = threadIdx.x & 31;
  const int m = L & 15, g = L >> 4;
  const _Float16* p = base + (size_t)(r0 + m) * ld + c0 + g * 8;
  v8h lo = *(const v8h*)p;
  v8h hi = *(const v8h*)(p + 16);
  v16h a;
#pragma unroll
  for (int j = 0; j < 8; ++j) { a[j] = lo[j]; a[j + 8] = hi[j]; }
  return a;
}

// B (32x16, KxN) from a [N][K]-major array T: lane L -> N=n0+(L&15), g=L>>4;
// b[0..7]=T[N][k0+g*8+j], b[8..15]=T[N][k0+16+g*8+j]
__device__ __forceinline__ v16h load_fragB(const _Float16* __restrict__ baseT,
                                           int ldt, int k0, int n0) {
  const int L = threadIdx.x & 31;
  const int n = L & 15, g = L >> 4;
  const _Float16* p = baseT + (size_t)(n0 + n) * ldt + k0 + g * 8;
  v8h lo = *(const v8h*)p;
  v8h hi = *(const v8h*)(p + 16);
  v16h b;
#pragma unroll
  for (int j = 0; j < 8; ++j) { b[j] = lo[j]; b[j + 8] = hi[j]; }
  return b;
}

// ---------------------------------------------------------------------------
// Conversion kernels
// ---------------------------------------------------------------------------
__global__ void cvt_f16_kernel(const float* __restrict__ in,
                               _Float16* __restrict__ out, int n) {
  int i = blockIdx.x * blockDim.x + threadIdx.x;
  if (i < n) out[i] = (_Float16)in[i];
}

// W [K x N] row-major f32  ->  WT [N x K] row-major f16
__global__ void cvt_wT_kernel(const float* __restrict__ W,
                              _Float16* __restrict__ WT, int K, int N) {
  int i = blockIdx.x * blockDim.x + threadIdx.x;
  if (i < K * N) {
    int k = i / N, n = i - k * N;
    WT[(size_t)n * K + k] = (_Float16)W[i];
  }
}

// ---------------------------------------------------------------------------
// QKV projection: out = relu(X @ W + b). One wave = 32x64 output tile,
// software-pipelined over K (next fragments loaded before current WMMAs).
// mode 0: store [B,H,L,DK] (Q,K); mode 1: store [B,H,DK,L] (V transposed)
// ---------------------------------------------------------------------------
__global__ void __launch_bounds__(256) proj_kernel(
    const _Float16* __restrict__ X, const _Float16* __restrict__ WT,
    const float* __restrict__ bias, _Float16* __restrict__ out,
    int Kdim, int mode) {
  const int wave = (int)((blockIdx.x * blockDim.x + threadIdx.x) >> 5);
  const int L = threadIdx.x & 31;
  const int r0 = (wave >> 3) * 32;          // 256 row tiles of 32
  const int n0 = (wave & 7) * 64;           // 8 col tiles of 64
  v8f accA[4] = {}, accB[4] = {};

  v16h a0 = load_fragA(X, Kdim, r0, 0);
  v16h a1 = load_fragA(X, Kdim, r0 + 16, 0);
  v16h b0 = load_fragB(WT, Kdim, 0, n0);
  v16h b1 = load_fragB(WT, Kdim, 0, n0 + 16);
  v16h b2 = load_fragB(WT, Kdim, 0, n0 + 32);
  v16h b3 = load_fragB(WT, Kdim, 0, n0 + 48);

  for (int k0 = 0; k0 < Kdim; k0 += 32) {
    const int kn = (k0 + 32 < Kdim) ? k0 + 32 : 0;   // wrap: harmless reload
    v16h na0 = load_fragA(X, Kdim, r0, kn);
    v16h na1 = load_fragA(X, Kdim, r0 + 16, kn);
    v16h nb0 = load_fragB(WT, Kdim, kn, n0);
    v16h nb1 = load_fragB(WT, Kdim, kn, n0 + 16);
    v16h nb2 = load_fragB(WT, Kdim, kn, n0 + 32);
    v16h nb3 = load_fragB(WT, Kdim, kn, n0 + 48);

    accA[0] = __builtin_amdgcn_wmma_f32_16x16x32_f16(false, a0, false, b0, (short)0, accA[0], false, false);
    accA[1] = __builtin_amdgcn_wmma_f32_16x16x32_f16(false, a0, false, b1, (short)0, accA[1], false, false);
    accA[2] = __builtin_amdgcn_wmma_f32_16x16x32_f16(false, a0, false, b2, (short)0, accA[2], false, false);
    accA[3] = __builtin_amdgcn_wmma_f32_16x16x32_f16(false, a0, false, b3, (short)0, accA[3], false, false);
    accB[0] = __builtin_amdgcn_wmma_f32_16x16x32_f16(false, a1, false, b0, (short)0, accB[0], false, false);
    accB[1] = __builtin_amdgcn_wmma_f32_16x16x32_f16(false, a1, false, b1, (short)0, accB[1], false, false);
    accB[2] = __builtin_amdgcn_wmma_f32_16x16x32_f16(false, a1, false, b2, (short)0, accB[2], false, false);
    accB[3] = __builtin_amdgcn_wmma_f32_16x16x32_f16(false, a1, false, b3, (short)0, accB[3], false, false);

    a0 = na0; a1 = na1; b0 = nb0; b1 = nb1; b2 = nb2; b3 = nb3;
  }

  const int mhalf = (L >> 4) * 8;
  const int nl = L & 15;
#pragma unroll
  for (int half = 0; half < 2; ++half) {
    const v8f* acc = half ? accB : accA;
#pragma unroll
    for (int t = 0; t < 4; ++t) {
      const int col = n0 + t * 16 + nl;
      const float bv = bias[col];
      const int h = col >> 6, dk = col & 63;
#pragma unroll
      for (int r = 0; r < 8; ++r) {
        const int row = r0 + half * 16 + mhalf + r;
        float v = acc[t][r] + bv;
        v = v > 0.f ? v : 0.f;
        const int bb = row >> 10, l = row & 1023;
        size_t idx = (mode == 0)
                         ? (((size_t)(bb * H_ + h) * LQ_ + l) * DK_ + dk)
                         : (((size_t)(bb * H_ + h) * DK_ + dk) * LK_ + l);
        out[idx] = (_Float16)v;
      }
    }
  }
}

// ---------------------------------------------------------------------------
// Flash attention, block-cooperative. Block = 8 waves, one (b,h), 8
// consecutive 16-query tiles. K/V 32-key chunks are async-staged into
// double-buffered LDS once per block; online softmax per wave.
// Qh,Kh: [B*H, L, 64] f16; Vt: [B*H, 64, L] f16; ctx: [B*L, 512] f16.
// ---------------------------------------------------------------------------
__global__ void __launch_bounds__(256) attn_kernel(
    const _Float16* __restrict__ Qh, const _Float16* __restrict__ Kh,
    const _Float16* __restrict__ Vt, const int* __restrict__ mask,
    _Float16* __restrict__ ctx) {
  __shared__ __align__(16) _Float16 sK[2][32 * 64];  // [key][dk]
  __shared__ __align__(16) _Float16 sV[2][64 * 32];  // [dk][key]
  __shared__ __align__(16) _Float16 sP[8][16 * 32];  // per-wave P staging

  const int tid = threadIdx.x;
  const int wslot = tid >> 5;
  const int L = tid & 31;
  const int bh = blockIdx.x >> 3;           // 64 (b,h) pairs
  const int qgrp = blockIdx.x & 7;          // 8 q-groups of 8 tiles
  const int b = bh >> 3, h = bh & 7;
  const int q0 = (qgrp * 8 + wslot) * 16;
  const _Float16* Qb = Qh + (size_t)bh * LQ_ * DK_;
  const _Float16* Kb = Kh + (size_t)bh * LK_ * DK_;
  const _Float16* Vb = Vt + (size_t)bh * DK_ * LK_;
  const int mhalf = (L >> 4) * 8;
  const int nl = L & 15;

  // Per-thread slices of the cooperative chunk copies (16 B each):
  // K chunk: 32 keys x 64 dk, contiguous 4 KB in Kb.
  const int kOff = tid * 8;                       // halves
  // V chunk: 64 dk-rows x 32 keys; global row stride 1024 halves.
  const int vRow = tid >> 2, vSeg = (tid & 3) * 8;

  // Q fragments reused across the whole key loop
  const v16h aq0 = load_fragA(Qb, DK_, q0, 0);
  const v16h aq1 = load_fragA(Qb, DK_, q0, 32);

  float m_i[8], l_i[8];
#pragma unroll
  for (int r = 0; r < 8; ++r) { m_i[r] = -3.0e38f; l_i[r] = 0.f; }
  v8f acc[4] = {};
  _Float16* sp = sP[wslot];

  // prologue: stage chunk 0 into buffer 0
  async_cp16(Kb + kOff, &sK[0][kOff]);
  async_cp16(Vb + (size_t)vRow * LK_ + vSeg, &sV[0][vRow * 32 + vSeg]);

  for (int kc = 0; kc < LK_ / 32; ++kc) {
    const int key0 = kc * 32;
    const int cur = kc & 1;
    const bool more = (kc + 1 < LK_ / 32);
    if (more) {  // stage next chunk into the other buffer
      const int key1 = key0 + 32;
      async_cp16(Kb + (size_t)key1 * DK_ + kOff, &sK[1 - cur][kOff]);
      async_cp16(Vb + (size_t)vRow * LK_ + key1 + vSeg,
                 &sV[1 - cur][vRow * 32 + vSeg]);
      WAIT_ASYNC_LE(2);   // current chunk (issued earlier) is complete
    } else {
      WAIT_ASYNC_LE(0);
    }
    __syncthreads();
    const _Float16* sKc = sK[cur];
    const _Float16* sVc = sV[cur];

    // ---- scores S[16q x 32k] from LDS K tile (contraction over dk)
    v8f s[2] = {};
#pragma unroll
    for (int t = 0; t < 2; ++t) {
      v16h bk0 = load_fragB(sKc, DK_, 0, t * 16);
      v16h bk1 = load_fragB(sKc, DK_, 32, t * 16);
      s[t] = __builtin_amdgcn_wmma_f32_16x16x32_f16(
          false, aq0, false, bk0, (short)0, s[t], false, false);
      s[t] = __builtin_amdgcn_wmma_f32_16x16x32_f16(
          false, aq1, false, bk1, (short)0, s[t], false, false);
    }
    // ---- scale + mask
#pragma unroll
    for (int t = 0; t < 2; ++t) {
      const int k = key0 + t * 16 + nl;
#pragma unroll
      for (int r = 0; r < 8; ++r) {
        const int q = q0 + mhalf + r;
        const int mv = mask[((size_t)b * LQ_ + q) * LK_ + k];
        const float sv = s[t][r] * 0.125f;   // 1/sqrt(64)
        s[t][r] = (mv != 0) ? sv : -1.0e8f;
      }
    }
    // ---- online softmax (row = 16 lanes of a half-wave, per VGPR slot)
#pragma unroll
    for (int r = 0; r < 8; ++r) {
      float cm = fmaxf(s[0][r], s[1][r]);
#pragma unroll
      for (int off = 1; off < 16; off <<= 1)
        cm = fmaxf(cm, __shfl_xor(cm, off, 32));
      const float mn = fmaxf(m_i[r], cm);
      const float p0 = __expf(s[0][r] - mn);
      const float p1 = __expf(s[1][r] - mn);
      s[0][r] = p0; s[1][r] = p1;
      float ls = p0 + p1;
#pragma unroll
      for (int off = 1; off < 16; off <<= 1)
        ls += __shfl_xor(ls, off, 32);
      const float alpha = __expf(m_i[r] - mn);
      l_i[r] = l_i[r] * alpha + ls;
      m_i[r] = mn;
#pragma unroll
      for (int t = 0; t < 4; ++t) acc[t][r] *= alpha;
    }
    // ---- transpose P (D-layout -> A-layout) through per-wave LDS
#pragma unroll
    for (int t = 0; t < 2; ++t)
#pragma unroll
      for (int r = 0; r < 8; ++r)
        sp[(mhalf + r) * 32 + t * 16 + nl] = (_Float16)s[t][r];
    v16h pa;
    {
      const _Float16* p = sp + nl * 32 + (L >> 4) * 8;
      v8h lo = *(const v8h*)p;
      v8h hi = *(const v8h*)(p + 16);
#pragma unroll
      for (int j = 0; j < 8; ++j) { pa[j] = lo[j]; pa[j + 8] = hi[j]; }
    }
    // ---- out += P @ V from LDS V tile (contraction over 32 keys)
#pragma unroll
    for (int t = 0; t < 4; ++t) {
      v16h bvf = load_fragB(sVc, 32, 0, t * 16);
      acc[t] = __builtin_amdgcn_wmma_f32_16x16x32_f16(
          false, pa, false, bvf, (short)0, acc[t], false, false);
    }
    __syncthreads();   // protect buffers before next overwrite
  }

  // ---- normalize + store context (heads concatenated: col = h*64 + dk)
#pragma unroll
  for (int t = 0; t < 4; ++t) {
#pragma unroll
    for (int r = 0; r < 8; ++r) {
      const int q = q0 + mhalf + r;
      const float ov = acc[t][r] / l_i[r];
      const int col = h * DK_ + t * 16 + nl;
      ctx[((size_t)b * LQ_ + q) * D_ + col] = (_Float16)ov;
    }
  }
}

// ---------------------------------------------------------------------------
// Output projection: out = ctx @ Wo + bo  (f32 result), 32x64 tile per wave
// ---------------------------------------------------------------------------
__global__ void __launch_bounds__(256) oproj_kernel(
    const _Float16* __restrict__ X, const _Float16* __restrict__ WT,
    const float* __restrict__ bias, float* __restrict__ out) {
  const int wave = (int)((blockIdx.x * blockDim.x + threadIdx.x) >> 5);
  const int L = threadIdx.x & 31;
  const int r0 = (wave >> 3) * 32;
  const int n0 = (wave & 7) * 64;
  v8f accA[4] = {}, accB[4] = {};

  v16h a0 = load_fragA(X, D_, r0, 0);
  v16h a1 = load_fragA(X, D_, r0 + 16, 0);
  v16h b0 = load_fragB(WT, D_, 0, n0);
  v16h b1 = load_fragB(WT, D_, 0, n0 + 16);
  v16h b2 = load_fragB(WT, D_, 0, n0 + 32);
  v16h b3 = load_fragB(WT, D_, 0, n0 + 48);

  for (int k0 = 0; k0 < D_; k0 += 32) {
    const int kn = (k0 + 32 < D_) ? k0 + 32 : 0;
    v16h na0 = load_fragA(X, D_, r0, kn);
    v16h na1 = load_fragA(X, D_, r0 + 16, kn);
    v16h nb0 = load_fragB(WT, D_, kn, n0);
    v16h nb1 = load_fragB(WT, D_, kn, n0 + 16);
    v16h nb2 = load_fragB(WT, D_, kn, n0 + 32);
    v16h nb3 = load_fragB(WT, D_, kn, n0 + 48);

    accA[0] = __builtin_amdgcn_wmma_f32_16x16x32_f16(false, a0, false, b0, (short)0, accA[0], false, false);
    accA[1] = __builtin_amdgcn_wmma_f32_16x16x32_f16(false, a0, false, b1, (short)0, accA[1], false, false);
    accA[2] = __builtin_amdgcn_wmma_f32_16x16x32_f16(false, a0, false, b2, (short)0, accA[2], false, false);
    accA[3] = __builtin_amdgcn_wmma_f32_16x16x32_f16(false, a0, false, b3, (short)0, accA[3], false, false);
    accB[0] = __builtin_amdgcn_wmma_f32_16x16x32_f16(false, a1, false, b0, (short)0, accB[0], false, false);
    accB[1] = __builtin_amdgcn_wmma_f32_16x16x32_f16(false, a1, false, b1, (short)0, accB[1], false, false);
    accB[2] = __builtin_amdgcn_wmma_f32_16x16x32_f16(false, a1, false, b2, (short)0, accB[2], false, false);
    accB[3] = __builtin_amdgcn_wmma_f32_16x16x32_f16(false, a1, false, b3, (short)0, accB[3], false, false);

    a0 = na0; a1 = na1; b0 = nb0; b1 = nb1; b2 = nb2; b3 = nb3;
  }

  const int mhalf = (L >> 4) * 8;
  const int nl = L & 15;
#pragma unroll
  for (int half = 0; half < 2; ++half) {
    const v8f* acc = half ? accB : accA;
#pragma unroll
    for (int t = 0; t < 4; ++t) {
      const int col = n0 + t * 16 + nl;
      const float bv = bias[col];
#pragma unroll
      for (int r = 0; r < 8; ++r) {
        const int row = r0 + half * 16 + mhalf + r;
        out[(size_t)row * D_ + col] = acc[t][r] + bv;
      }
    }
  }
}

// ---------------------------------------------------------------------------
// Host launcher
// ---------------------------------------------------------------------------
extern "C" void kernel_launch(void* const* d_in, const int* in_sizes, int n_in,
                              void* d_out, int out_size, void* d_ws, size_t ws_size,
                              hipStream_t stream) {
  const float* query = (const float*)d_in[0];
  const float* keyi  = (const float*)d_in[1];
  const int*   mask  = (const int*)d_in[2];
  const float* Wq = (const float*)d_in[3];
  const float* bq = (const float*)d_in[4];
  const float* Wk = (const float*)d_in[5];
  const float* bk = (const float*)d_in[6];
  const float* Wv = (const float*)d_in[7];
  const float* bv = (const float*)d_in[8];
  const float* Wo = (const float*)d_in[9];
  const float* bo = (const float*)d_in[10];
  float* out = (float*)d_out;

  char* ws = (char*)d_ws;
  size_t off = 0;
  auto alloc = [&](size_t bytes) {
    char* p = ws + off;
    off = (off + bytes + 255) & ~(size_t)255;
    return p;
  };
  _Float16* q16 = (_Float16*)alloc((size_t)MROWS * QIN_ * 2);
  _Float16* k16 = (_Float16*)alloc((size_t)MROWS * QIN_ * 2);
  _Float16* WqT = (_Float16*)alloc((size_t)D_ * QIN_ * 2);
  _Float16* WkT = (_Float16*)alloc((size_t)D_ * QIN_ * 2);
  _Float16* WvT = (_Float16*)alloc((size_t)D_ * QIN_ * 2);
  _Float16* WoT = (_Float16*)alloc((size_t)D_ * D_ * 2);
  _Float16* Qh  = (_Float16*)alloc((size_t)B_ * H_ * LQ_ * DK_ * 2);
  _Float16* Kh  = (_Float16*)alloc((size_t)B_ * H_ * LK_ * DK_ * 2);
  _Float16* Vt  = (_Float16*)alloc((size_t)B_ * H_ * DK_ * LK_ * 2);
  _Float16* ctx = (_Float16*)alloc((size_t)MROWS * D_ * 2);
  (void)ws_size; (void)n_in; (void)in_sizes; (void)out_size;

  const int nX = MROWS * QIN_;
  cvt_f16_kernel<<<(nX + 255) / 256, 256, 0, stream>>>(query, q16, nX);
  cvt_f16_kernel<<<(nX + 255) / 256, 256, 0, stream>>>(keyi, k16, nX);
  cvt_wT_kernel<<<(QIN_ * D_ + 255) / 256, 256, 0, stream>>>(Wq, WqT, QIN_, D_);
  cvt_wT_kernel<<<(QIN_ * D_ + 255) / 256, 256, 0, stream>>>(Wk, WkT, QIN_, D_);
  cvt_wT_kernel<<<(QIN_ * D_ + 255) / 256, 256, 0, stream>>>(Wv, WvT, QIN_, D_);
  cvt_wT_kernel<<<(D_ * D_ + 255) / 256, 256, 0, stream>>>(Wo, WoT, D_, D_);

  // 32x64 tiles: (8192/32)*(512/64) = 2048 waves -> 256 blocks of 256 threads
  const int gemmBlocks = (MROWS / 32) * (D_ / 64) * 32 / 256;  // 256
  proj_kernel<<<gemmBlocks, 256, 0, stream>>>(q16, WqT, bq, Qh, QIN_, 0);
  proj_kernel<<<gemmBlocks, 256, 0, stream>>>(k16, WkT, bk, Kh, QIN_, 0);
  proj_kernel<<<gemmBlocks, 256, 0, stream>>>(k16, WvT, bv, Vt, QIN_, 1);

  // 64 (b,h) * 8 q-groups = 512 blocks; 8 waves/block share K/V via LDS
  attn_kernel<<<512, 256, 0, stream>>>(Qh, Kh, Vt, mask, ctx);

  oproj_kernel<<<gemmBlocks, 256, 0, stream>>>(ctx, WoT, bo, out);
}
